// Network_10909216932570
// MI455X (gfx1250) — matrix operand order repaired and checked by reading, not compiled, not run on hardware
//
#include <hip/hip_runtime.h>
#include <hip/hip_bf16.h>
#include <cstddef>

// Problem constants
#define BB    32
#define SEQ   512
#define FEAT  768
#define DMAX  48
#define CMAX  64
#define KH    4
#define GOUT  192
#define ROWS  (BB * DMAX)      // 1536

typedef __attribute__((ext_vector_type(16))) __bf16 v16bf;
typedef __attribute__((ext_vector_type(8)))  __bf16 v8bf;
typedef __attribute__((ext_vector_type(8)))  float  v8f;
typedef __attribute__((ext_vector_type(4)))  float  v4f;

// ---------------------------------------------------------------------------
// K0: build swizzled bf16 B matrix (FEAT x KH*GOUT = 768x768).
// Element e = ((ktile*48 + ntile)*32 + lane)*16 + idx holds
//   B[k = ktile*32 + (lane/16)*16 + idx, n = ntile*16 + lane%16]
// so each WMMA lane loads its 16 B values as two contiguous 16B loads.
// ---------------------------------------------------------------------------
__global__ __launch_bounds__(256) void k0_prep(const float* __restrict__ gatW,
                                               __bf16* __restrict__ Bswz) {
    const int e = blockIdx.x * 256 + threadIdx.x;
    if (e >= FEAT * FEAT) return;
    const int idx  = e & 15;
    const int lane = (e >> 4) & 31;
    const int nt   = (e >> 9) % 48;
    const int kt   = (e >> 9) / 48;
    const int kk   = kt * 32 + ((lane >> 4) << 4) + idx;
    const int nn   = nt * 16 + (lane & 15);
    const int head = nn / GOUT, o = nn % GOUT;
    Bswz[e] = (__bf16)gatW[((size_t)head * FEAT + kk) * GOUT + o];
}

// ---------------------------------------------------------------------------
// K1: sentence pooling. One block per (b, d) clause. Only touches the
// clause_len (<=10) live tokens: masked tokens contribute exactly zero in
// the reference (softmax weight underflows to 0 AND sent row is zeroed).
// ---------------------------------------------------------------------------
__global__ __launch_bounds__(256) void k1_sent(const float* __restrict__ hs,
                                               const float* __restrict__ fc_w,
                                               const float* __restrict__ fc_b,
                                               const int*   __restrict__ clen,
                                               const int*   __restrict__ dlen,
                                               float*  __restrict__ doch,
                                               __bf16* __restrict__ docA) {
    const int bd   = blockIdx.x;
    const int b    = bd / DMAX, d = bd % DMAX;
    const int tid  = threadIdx.x, lane = tid & 31, wave = tid >> 5;
    __shared__ float sc[16];
    __shared__ float alpha[16];
    __shared__ int   sidx[16];

    int start = 0;
    for (int i = 0; i < d; ++i) start += clen[b * DMAX + i];
    const int cl   = clen[b * DMAX + d];                // 1..10
    const int nval = (d < dlen[b]) ? min(cl, 16) : 0;

    // token scores: one wave per live token (c = wave, wave+8, ...)
    for (int c = wave; c < nval; c += 8) {
        const int idx = min(start + c, SEQ - 1);
        float acc = 0.f;
        const float* rowp = hs + ((size_t)b * SEQ + idx) * FEAT;
        for (int f = lane; f < FEAT; f += 32) acc += rowp[f] * fc_w[f];
        for (int off = 16; off > 0; off >>= 1) acc += __shfl_down(acc, off, 32);
        if (lane == 0) { sc[c] = acc + fc_b[0]; sidx[c] = idx; }
    }
    __syncthreads();

    if (tid == 0 && nval > 0) {                         // softmax over live tokens
        float mx = -1e30f;
        for (int c = 0; c < nval; ++c) mx = fmaxf(mx, sc[c]);
        float s = 0.f;
        for (int c = 0; c < nval; ++c) { float e = __expf(sc[c] - mx); alpha[c] = e; s += e; }
        const float inv = 1.f / s;
        for (int c = 0; c < nval; ++c) alpha[c] *= inv;
    }
    __syncthreads();

    // weighted sum over live tokens -> fp32 + bf16 (WMMA A) copies
    for (int f = tid; f < FEAT; f += 256) {
        float acc = 0.f;
        for (int c = 0; c < nval; ++c)
            acc += alpha[c] * hs[((size_t)b * SEQ + sidx[c]) * FEAT + f];
        doch[(size_t)bd * FEAT + f] = acc;
        docA[(size_t)bd * FEAT + f] = (__bf16)acc;
    }
}

// ---------------------------------------------------------------------------
// K2: h_prime = A(1536x768,bf16) @ Bswz(768x768,bf16) via WMMA bf16.
// One wave per 32x32 output block (2x2 register tiling): 2 A-loads + 2
// B-loads feed 4 v_wmma_f32_16x16x32_bf16 per K-step; 24 K-steps.
// ---------------------------------------------------------------------------
__device__ __forceinline__ v16bf ld_bf16x16(const __bf16* p0, const __bf16* p1) {
    v8bf lo = *reinterpret_cast<const v8bf*>(p0);
    v8bf hi = *reinterpret_cast<const v8bf*>(p1);
    v16bf r;
#pragma unroll
    for (int i = 0; i < 8; ++i) { r[i] = lo[i]; r[8 + i] = hi[i]; }
    return r;
}

__device__ __forceinline__ void st_tile(float* __restrict__ hp, const v8f& c,
                                        int mt, int nt, int lane) {
    // C layout: VGPR r, lane l -> row = r + 8*(l/16), col = l%16
    const int colg  = nt * 16 + (lane & 15);
    const int rbase = mt * 16 + ((lane >> 4) << 3);
#pragma unroll
    for (int r = 0; r < 8; ++r)
        hp[(size_t)(rbase + r) * FEAT + colg] = c[r];
}

__global__ __launch_bounds__(256) void k2_gemm(const __bf16* __restrict__ A,
                                               const __bf16* __restrict__ Bswz,
                                               float* __restrict__ hp) {
    const int lane  = threadIdx.x & 31;
    const int wv    = blockIdx.x * 8 + (threadIdx.x >> 5);  // 0..1151
    const int mg    = wv / 24;                              // 0..47 (M groups of 32)
    const int ng    = wv % 24;                              // 0..23 (N groups of 32)
    const int mt0   = mg * 2, nt0 = ng * 2;

    const int rowA0 = mt0 * 16 + (lane & 15);
    const int koffA = (lane >> 4) * 16;                     // lanes 16-31: K+16

    v8f c00 = {}, c01 = {}, c10 = {}, c11 = {};
#pragma unroll 2
    for (int kt = 0; kt < FEAT / 32; ++kt) {
        const __bf16* ap0 = A + (size_t)rowA0 * FEAT + kt * 32 + koffA;
        const __bf16* ap1 = ap0 + (size_t)16 * FEAT;        // next 16 rows
        const __bf16* bp0 = Bswz + (((size_t)kt * 48 + nt0) * 32 + lane) * 16;
        const __bf16* bp1 = bp0 + 512;                      // next N tile
        v16bf a0 = ld_bf16x16(ap0, ap0 + 8);
        v16bf a1 = ld_bf16x16(ap1, ap1 + 8);
        v16bf b0 = ld_bf16x16(bp0, bp0 + 8);
        v16bf b1 = ld_bf16x16(bp1, bp1 + 8);
        c00 = __builtin_amdgcn_wmma_f32_16x16x32_bf16(false, a0, false, b0, (short)0, c00, false, false);
        c01 = __builtin_amdgcn_wmma_f32_16x16x32_bf16(false, a0, false, b1, (short)0, c01, false, false);
        c10 = __builtin_amdgcn_wmma_f32_16x16x32_bf16(false, a1, false, b0, (short)0, c10, false, false);
        c11 = __builtin_amdgcn_wmma_f32_16x16x32_bf16(false, a1, false, b1, (short)0, c11, false, false);
    }
    st_tile(hp, c00, mt0,     nt0,     lane);
    st_tile(hp, c01, mt0,     nt0 + 1, lane);
    st_tile(hp, c10, mt0 + 1, nt0,     lane);
    st_tile(hp, c11, mt0 + 1, nt0 + 1, lane);
}

// ---------------------------------------------------------------------------
// K3: per-(b,k) GAT attention. Stage the 48x192 h_prime block in LDS once
// (36.9KB; read ~50x by tanh pass + attn@h' matmul), then:
// tanh-projected src/dst scores, leaky-relu, adj mask, row softmax,
// feat = attn @ h_prime. Deterministic (no atomics).
// ---------------------------------------------------------------------------
__global__ __launch_bounds__(256) void k3_attn(const float* __restrict__ hp,
                                               const float* __restrict__ wsrc,
                                               const float* __restrict__ wdst,
                                               const int*   __restrict__ adj,
                                               float* __restrict__ featb) {
    const int bk = blockIdx.x;                    // b*KH + k
    const int b = bk / KH, k = bk % KH;
    const int tid = threadIdx.x;
    __shared__ float hps[DMAX][GOUT];             // 36864 B
    __shared__ float attn[DMAX][DMAX];            //  9216 B
    __shared__ float asrc[DMAX], adst[DMAX];

    // stage h_prime block (vectorized float4 loads)
    for (int e = tid; e < DMAX * (GOUT / 4); e += 256) {
        const int n = e / (GOUT / 4), q = e % (GOUT / 4);
        v4f v = *reinterpret_cast<const v4f*>(
            hp + (size_t)(b * DMAX + n) * FEAT + k * GOUT + q * 4);
        *reinterpret_cast<v4f*>(&hps[n][q * 4]) = v;
    }
    __syncthreads();

    if (tid < 2 * DMAX) {                         // a_src / a_dst over tanh(h')
        const int n = tid >> 1;
        const int isdst = tid & 1;
        const float* w = (isdst ? wdst : wsrc) + k * GOUT;
        float s = 0.f;
        for (int o = 0; o < GOUT; ++o) s += tanhf(hps[n][o]) * w[o];
        if (isdst) adst[n] = s; else asrc[n] = s;
    }
    __syncthreads();

    for (int e = tid; e < DMAX * DMAX; e += 256) {
        const int n = e / DMAX, m = e % DMAX;
        float x = asrc[n] + adst[m];
        x = (x >= 0.f) ? x : 0.2f * x;            // leaky_relu(0.2)
        if (adj[((size_t)b * DMAX + n) * DMAX + m] == 0) x = -999.0f;
        attn[n][m] = x;
    }
    __syncthreads();

    if (tid < DMAX) {                             // row softmax
        float mx = -1e30f;
        for (int m = 0; m < DMAX; ++m) mx = fmaxf(mx, attn[tid][m]);
        float s = 0.f;
        for (int m = 0; m < DMAX; ++m) { float e = __expf(attn[tid][m] - mx); attn[tid][m] = e; s += e; }
        const float inv = 1.f / s;
        for (int m = 0; m < DMAX; ++m) attn[tid][m] *= inv;
    }
    __syncthreads();

    for (int e = tid; e < DMAX * GOUT; e += 256) { // feat = attn @ h_prime
        const int n = e / GOUT, o = e % GOUT;
        float acc = 0.f;
#pragma unroll 4
        for (int m = 0; m < DMAX; ++m)
            acc += attn[n][m] * hps[m][o];
        featb[((size_t)bk * DMAX + n) * GOUT + o] = acc;
    }
}

// ---------------------------------------------------------------------------
// K4: bias + residual + elu, dot with out_emo_w, sigmoid -> d_out (1536).
// One block per (b, n) row.
// ---------------------------------------------------------------------------
__global__ __launch_bounds__(256) void k4_out(const float* __restrict__ featb,
                                              const float* __restrict__ gat_b,
                                              const float* __restrict__ doch,
                                              const float* __restrict__ w_out,
                                              const float* __restrict__ b_out,
                                              float* __restrict__ out) {
    const int row = blockIdx.x;                   // b*DMAX + n
    const int b = row / DMAX, n = row % DMAX;
    const int tid = threadIdx.x;
    float p = 0.f;
    for (int f = tid; f < FEAT; f += 256) {
        const int k = f / GOUT, o = f % GOUT;
        float x = featb[((size_t)(b * KH + k) * DMAX + n) * GOUT + o]
                + gat_b[o] + doch[(size_t)row * FEAT + f];
        x = (x > 0.f) ? x : (__expf(x) - 1.f);    // elu
        p += x * w_out[f];
    }
    __shared__ float red[256];
    red[tid] = p; __syncthreads();
    for (int s = 128; s > 0; s >>= 1) {
        if (tid < s) red[tid] += red[tid + s];
        __syncthreads();
    }
    if (tid == 0) out[row] = 1.f / (1.f + __expf(-(red[0] + b_out[0])));
}

// ---------------------------------------------------------------------------
extern "C" void kernel_launch(void* const* d_in, const int* in_sizes, int n_in,
                              void* d_out, int out_size, void* d_ws, size_t ws_size,
                              hipStream_t stream) {
    const float* hs       = (const float*)d_in[0];   // (32,512,768)
    const float* fc_w     = (const float*)d_in[1];   // (768,1)
    const float* fc_b     = (const float*)d_in[2];   // (1,)
    const float* gat_W    = (const float*)d_in[3];   // (4,768,192)
    const float* gat_b    = (const float*)d_in[4];   // (192,)
    const float* gat_wsrc = (const float*)d_in[5];   // (4,192,1)
    const float* gat_wdst = (const float*)d_in[6];   // (4,192,1)
    const float* out_w    = (const float*)d_in[7];   // (768,1)
    const float* out_b    = (const float*)d_in[8];   // (1,)
    const int*   clen     = (const int*)d_in[9];     // (32,48)
    const int*   dlen     = (const int*)d_in[10];    // (32,)
    const int*   adj      = (const int*)d_in[11];    // (32,48,48)
    float* out = (float*)d_out;

    char* ws = (char*)d_ws;
    float*  doch  = (float*)(ws);                          // 1536*768*4 = 4,718,592
    __bf16* docA  = (__bf16*)(ws + 4718592);               // 1536*768*2 = 2,359,296
    __bf16* Bswz  = (__bf16*)(ws + 7077888);               //  768*768*2 = 1,179,648
    float*  hp    = (float*)(ws + 8257536);                // 1536*768*4 = 4,718,592
    float*  featb = (float*)(ws + 12976128);               // 1536*768*4 = 4,718,592
                                                           // total ~17.7 MB

    k0_prep<<<(FEAT * FEAT + 255) / 256, 256, 0, stream>>>(gat_W, Bswz);
    k1_sent<<<ROWS, 256, 0, stream>>>(hs, fc_w, fc_b, clen, dlen, doch, docA);
    k2_gemm<<<(ROWS / 32) * (FEAT / 32) / 8, 256, 0, stream>>>(docA, Bswz, hp);
    k3_attn<<<BB * KH, 256, 0, stream>>>(hp, gat_wsrc, gat_wdst, adj, featb);
    k4_out<<<ROWS, 256, 0, stream>>>(featb, gat_b, doch, out_w, out_b, out);
}
